// ResnetBlockConvReluLattice_28664611733902
// MI455X (gfx1250) — matrix operand order repaired and checked by reading, not compiled, not run on hardware
//
#include <hip/hip_runtime.h>
#include <hip/hip_bf16.h>
#include <stddef.h>

// ---------------------------------------------------------------------------
// ResnetBlockConvReluLattice on gfx1250 (MI455X), bf16 WMMA path.
//   layer: out[N,64] = relu( gather(x, neigh)[N, 9*64] @ W[9*64, 64] + b )
//   block: h = layer1(lv); out = layer2(h) + lv
// ---------------------------------------------------------------------------

#define LAT_N  200000
#define LAT_C  64
#define LAT_K  9
#define LAT_KC (LAT_K * LAT_C)   // 576
#define KCHUNKS (LAT_KC / 32)    // 18 chunks of K=32 for wmma 16x16x32

typedef __attribute__((ext_vector_type(16))) __bf16 v16bf;
typedef __attribute__((ext_vector_type(8)))  __bf16 v8bf;
typedef __attribute__((ext_vector_type(8)))  float  v8f;

// Build a 16-element bf16 fragment from two contiguous 8-element (16B) chunks.
__device__ __forceinline__ v16bf load_frag(const __bf16* p, int second_off) {
    v8bf lo = *(const v8bf*)(p);
    v8bf hi = *(const v8bf*)(p + second_off);
    return __builtin_shufflevector(lo, hi,
        0, 1, 2, 3, 4, 5, 6, 7, 8, 9, 10, 11, 12, 13, 14, 15);
}

// -------------------------- prep kernels -----------------------------------

// fp32 -> bf16 elementwise (grid-stride)
__global__ void lat_cvt_bf16(const float* __restrict__ src,
                             __bf16* __restrict__ dst, int n) {
    for (int i = blockIdx.x * blockDim.x + threadIdx.x; i < n;
         i += gridDim.x * blockDim.x) {
        dst[i] = (__bf16)src[i];
    }
}

// w[576][64] fp32 (row-major, K-major) -> wt[64][576] bf16 (N-major rows)
__global__ void lat_prep_w(const float* __restrict__ w,
                           __bf16* __restrict__ wt) {
    int i = blockIdx.x * blockDim.x + threadIdx.x;
    if (i < LAT_KC * LAT_C) {
        int k = i / LAT_C;
        int n = i % LAT_C;
        wt[n * LAT_KC + k] = (__bf16)w[i];
    }
}

// -------------------------- main conv kernel -------------------------------
// One wave (32 lanes) computes a 16-vertex x 64-channel output tile.
// A: gathered neighbor features (bf16), fragments loaded straight from global.
// B: pre-transposed weights wt[64][576] bf16, contiguous per-lane fragments.
// Accumulate f32 via v_wmma_f32_16x16x32_bf16; 18 K-chunks x 4 N-tiles.
template <bool RESIDUAL>
__global__ __launch_bounds__(256)
void lat_conv_wmma(const __bf16* __restrict__ x,     // [N,64] bf16 features
                   const int*    __restrict__ neigh, // [N,9]
                   const __bf16* __restrict__ wt,    // [64][576] bf16
                   const float*  __restrict__ bias,  // [64]
                   const float*  __restrict__ ident, // [N,64] f32 (RESIDUAL)
                   __bf16*       __restrict__ out_bf,// layer-1 output
                   float*        __restrict__ out_f) // layer-2 output
{
    const int wave = blockIdx.x * (blockDim.x >> 5) + (threadIdx.x >> 5);
    const int m0   = wave * 16;
    if (m0 >= LAT_N) return;                 // wave-uniform early out

    const int lane = threadIdx.x & 31;
    const int r    = lane & 15;              // A row / B column within tile
    const int half = lane >> 4;              // K-half selector
    const int vrow = m0 + r;

    v8f acc0 = {}, acc1 = {}, acc2 = {}, acc3 = {};

#pragma unroll
    for (int kc = 0; kc < KCHUNKS; ++kc) {
        // K-chunk kc covers k = kc*32 .. kc*32+31, all inside neighbor nb.
        const int nb  = kc >> 1;
        const int idx = neigh[vrow * LAT_K + nb];

        // A fragment: row r, K = {half*8..+7, 16+half*8..+7} of this chunk.
        const __bf16* ap =
            x + (size_t)idx * LAT_C + (kc & 1) * 32 + half * 8;
        v16bf a = load_frag(ap, 16);

        // B fragments: column nt*16+r, K = {half*8..+7, 16+half*8..+7} of this chunk.
        const int koff = kc * 32 + half * 8;
        const __bf16* wrow = wt + (size_t)r * LAT_KC + koff;
        v16bf b0 = load_frag(wrow + (size_t)(0 * 16) * LAT_KC, 16);
        v16bf b1 = load_frag(wrow + (size_t)(1 * 16) * LAT_KC, 16);
        v16bf b2 = load_frag(wrow + (size_t)(2 * 16) * LAT_KC, 16);
        v16bf b3 = load_frag(wrow + (size_t)(3 * 16) * LAT_KC, 16);

        acc0 = __builtin_amdgcn_wmma_f32_16x16x32_bf16(
            false, a, false, b0, (short)0, acc0, false, false);
        acc1 = __builtin_amdgcn_wmma_f32_16x16x32_bf16(
            false, a, false, b1, (short)0, acc1, false, false);
        acc2 = __builtin_amdgcn_wmma_f32_16x16x32_bf16(
            false, a, false, b2, (short)0, acc2, false, false);
        acc3 = __builtin_amdgcn_wmma_f32_16x16x32_bf16(
            false, a, false, b3, (short)0, acc3, false, false);
    }

    // Epilogue: D layout -> lane holds col = nt*16 + r, rows v + half*8.
    v8f accs[4] = {acc0, acc1, acc2, acc3};
#pragma unroll
    for (int nt = 0; nt < 4; ++nt) {
        const int col = nt * 16 + r;
        const float bv = bias[col];
#pragma unroll
        for (int v = 0; v < 8; ++v) {
            const int row = v + half * 8;
            float val = accs[nt][v] + bv;
            val = fmaxf(val, 0.0f);
            const size_t o = (size_t)(m0 + row) * LAT_C + col;
            if (RESIDUAL) {
                out_f[o] = val + ident[o];
            } else {
                out_bf[o] = (__bf16)val;
            }
        }
    }
}

// -------------------------- host launcher ----------------------------------

extern "C" void kernel_launch(void* const* d_in, const int* in_sizes, int n_in,
                              void* d_out, int out_size, void* d_ws, size_t ws_size,
                              hipStream_t stream) {
    (void)in_sizes; (void)n_in; (void)out_size; (void)ws_size;

    const float* lv     = (const float*)d_in[0];
    const int*   neigh1 = (const int*)  d_in[1];
    const int*   neigh2 = (const int*)  d_in[2];
    const float* w1     = (const float*)d_in[3];
    const float* b1     = (const float*)d_in[4];
    const float* w2     = (const float*)d_in[5];
    const float* b2     = (const float*)d_in[6];
    float* out = (float*)d_out;

    // Workspace layout (bytes):
    //   wt1   : 576*64 bf16                    =    73,728
    //   wt2   : 576*64 bf16                    =    73,728
    //   lv_bf : N*64 bf16                      = 25,600,000
    //   h_bf  : N*64 bf16                      = 25,600,000
    char* ws = (char*)d_ws;
    __bf16* wt1   = (__bf16*)(ws);
    __bf16* wt2   = (__bf16*)(ws + 73728);
    __bf16* lv_bf = (__bf16*)(ws + 2 * 73728);
    __bf16* h_bf  = (__bf16*)(ws + 2 * 73728 + (size_t)LAT_N * LAT_C * 2);

    // 1) precision prep: fp32 -> bf16 feature table, transposed bf16 weights
    lat_cvt_bf16<<<4096, 256, 0, stream>>>(lv, lv_bf, LAT_N * LAT_C);
    lat_prep_w<<<(LAT_KC * LAT_C + 255) / 256, 256, 0, stream>>>(w1, wt1);
    lat_prep_w<<<(LAT_KC * LAT_C + 255) / 256, 256, 0, stream>>>(w2, wt2);

    // 2) two conv layers: 12500 wave-tiles, 8 waves (256 threads) per block
    const int tiles  = LAT_N / 16;                 // 12500
    const int blocks = (tiles + 7) / 8;            // 1563

    lat_conv_wmma<false><<<blocks, 256, 0, stream>>>(
        lv_bf, neigh1, wt1, b1, /*ident=*/nullptr, h_bf, /*out_f=*/nullptr);

    lat_conv_wmma<true><<<blocks, 256, 0, stream>>>(
        h_bf, neigh2, wt2, b2, /*ident=*/lv, /*out_bf=*/nullptr, out);
}
